// SAGE_6451040879152
// MI455X (gfx1250) — compile-verified, matrix operand order checked by reading, not run on hardware
//
#include <hip/hip_runtime.h>
#include <hip/hip_bf16.h>
#include <stdint.h>

// ---------------------------------------------------------------------------
// GraphSAGE 2-layer forward for MI455X (gfx1250, wave32, WMMA).
// GEMMs fused per layer via K-concatenation; each wave computes a 64x16 block
// (4 M-tiles sharing one B fragment) so 4 independent v_wmma ops issue per
// K-step, keeping the matrix pipe busy across load latency.
// ---------------------------------------------------------------------------

typedef __attribute__((ext_vector_type(16))) __bf16 bf16x16;
typedef __attribute__((ext_vector_type(8)))  __bf16 bf16x8;
typedef __attribute__((ext_vector_type(8)))  float  f32x8;

#define GN2   180224
#define GN1   11264
#define GN0   1024
#define GE1   168960
#define GE2   10240
#define CIN   602
#define HID   256
#define OUTC  41
#define K1    1216      // 2*602 = 1204 padded to 38*32
#define K2    512       // 2*256, = 16*32
#define NPAD2 48        // 41 padded to 3*16

// round-to-nearest-even f32 -> bf16 (bit-exact, no reliance on __bf16 arith)
__device__ __forceinline__ __bf16 f2bf(float f) {
    union { float f; uint32_t u; } v; v.f = f;
    uint32_t r = (v.u + 0x7FFFu + ((v.u >> 16) & 1u)) >> 16;
    union { unsigned short s; __bf16 b; } o; o.s = (unsigned short)r;
    return o.b;
}

// Load one 16-lane-striped bf16 WMMA fragment chunk pair (2 x b128 per lane).
__device__ __forceinline__ bf16x16 load_frag(const __bf16* p) {
    union { bf16x16 v; bf16x8 h[2]; } u;
    u.h[0] = *(const bf16x8*)(p);
    u.h[1] = *(const bf16x8*)(p + 16);
    return u.v;
}

// ---------------------------------------------------------------------------
__global__ void zero_f32(float* __restrict__ p, long n) {
    long i = (long)blockIdx.x * blockDim.x + threadIdx.x;
    long stride = (long)gridDim.x * blockDim.x;
    for (; i < n; i += stride) p[i] = 0.0f;
}

// Pack [W1l;W1r] (each 602x256, row-major) into transposed bf16 [256][1216].
__global__ void pack_w1(const float* __restrict__ Wl, const float* __restrict__ Wr,
                        __bf16* __restrict__ WT) {
    int idx = blockIdx.x * blockDim.x + threadIdx.x;
    if (idx >= HID * K1) return;
    int n = idx / K1, k = idx - n * K1;
    float v = 0.0f;
    if (k < CIN)          v = Wl[(size_t)k * HID + n];
    else if (k < 2 * CIN) v = Wr[(size_t)(k - CIN) * HID + n];
    WT[idx] = f2bf(v);
}

// Pack [W2l;W2r] (each 256x41) into transposed bf16 [48][512]; pad cols -> 0.
__global__ void pack_w2(const float* __restrict__ Wl, const float* __restrict__ Wr,
                        __bf16* __restrict__ WT) {
    int idx = blockIdx.x * blockDim.x + threadIdx.x;
    if (idx >= NPAD2 * K2) return;
    int n = idx / K2, k = idx - n * K2;
    float v = 0.0f;
    if (n < OUTC) v = (k < HID) ? Wl[(size_t)k * OUTC + n]
                                : Wr[(size_t)(k - HID) * OUTC + n];
    WT[idx] = f2bf(v);
}

// One block per edge: atomic f32 segment-sum of source features + edge count.
__global__ __launch_bounds__(128) void scatter_mean(
    const float* __restrict__ feat, const int* __restrict__ src,
    const int* __restrict__ dst, float* __restrict__ sum,
    float* __restrict__ cnt, int C) {
    int e = blockIdx.x;
    int s = src[e], d = dst[e];
    const float* fs = feat + (size_t)s * C;
    float*       sd = sum  + (size_t)d * C;
    for (int c = threadIdx.x; c < C; c += blockDim.x)
        atomicAdd(&sd[c], fs[c]);
    if (threadIdx.x == 0) atomicAdd(&cnt[d], 1.0f);
}

// Row r of A1 = [ sum1[r,:]/max(cnt,1) | x[r,:] | 0-pad ]  as bf16 [11264][1216]
__global__ __launch_bounds__(256) void build_a1(
    const float* __restrict__ sum1, const float* __restrict__ cnt1,
    const float* __restrict__ x, __bf16* __restrict__ A1) {
    int r = blockIdx.x;
    float inv = 1.0f / fmaxf(cnt1[r], 1.0f);
    const float* srow = sum1 + (size_t)r * CIN;
    const float* xrow = x    + (size_t)r * CIN;
    __bf16* arow = A1 + (size_t)r * K1;
    for (int k = threadIdx.x; k < K1; k += blockDim.x) {
        float v = (k < CIN) ? srow[k] * inv
                            : ((k < 2 * CIN) ? xrow[k - CIN] : 0.0f);
        arow[k] = f2bf(v);
    }
}

// Row r of A2 = [ sum2[r,:]/max(cnt,1) | h[r,:] ]  as bf16 [1024][512]
__global__ __launch_bounds__(256) void build_a2(
    const float* __restrict__ sum2, const float* __restrict__ cnt2,
    const float* __restrict__ H, __bf16* __restrict__ A2) {
    int r = blockIdx.x;
    float inv = 1.0f / fmaxf(cnt2[r], 1.0f);
    const float* srow = sum2 + (size_t)r * HID;
    const float* hrow = H    + (size_t)r * HID;
    __bf16* arow = A2 + (size_t)r * K2;
    for (int k = threadIdx.x; k < K2; k += blockDim.x) {
        float v = (k < HID) ? srow[k] * inv : hrow[k - HID];
        arow[k] = f2bf(v);
    }
}

// H[11264,256] = ReLU(A1[11264,1216] @ W^T + b1).
// Wave tile: 64 rows x 16 cols = 4 M-tiles sharing one B fragment per K-step.
// Fragment layout per ISA 7.12.2: lane l holds row (l&15); chunks at
// K-offsets (l>>4)*8 and (l>>4)*8+16 within each 32-wide K step.
__global__ __launch_bounds__(128) void gemm1_wmma(
    const __bf16* __restrict__ A, const __bf16* __restrict__ BT,
    const float* __restrict__ bias, float* __restrict__ H) {
    const int lane = threadIdx.x & 31;
    const int wave = threadIdx.x >> 5;
    const int m0 = blockIdx.x * 64;
    const int n0 = (blockIdx.y * 4 + wave) * 16;
    const int row = lane & 15;
    const int hi  = lane >> 4;
    const __bf16* aptr = A  + (size_t)(m0 + row) * K1 + hi * 8;
    const __bf16* bptr = BT + (size_t)(n0 + row) * K1 + hi * 8;
    const f32x8 z = {0.f, 0.f, 0.f, 0.f, 0.f, 0.f, 0.f, 0.f};
    f32x8 acc[4] = {z, z, z, z};
    for (int kk = 0; kk < K1 / 32; ++kk) {
        const int ko = kk * 32;
        __builtin_prefetch(aptr + ko + 128, 0, 1);    // speculative, safe OOB
        bf16x16 b = load_frag(bptr + ko);
        bf16x16 a0 = load_frag(aptr + ko);
        bf16x16 a1 = load_frag(aptr + ko + (size_t)16 * K1);
        bf16x16 a2 = load_frag(aptr + ko + (size_t)32 * K1);
        bf16x16 a3 = load_frag(aptr + ko + (size_t)48 * K1);
        acc[0] = __builtin_amdgcn_wmma_f32_16x16x32_bf16(
            false, a0, false, b, (short)0, acc[0], false, false);
        acc[1] = __builtin_amdgcn_wmma_f32_16x16x32_bf16(
            false, a1, false, b, (short)0, acc[1], false, false);
        acc[2] = __builtin_amdgcn_wmma_f32_16x16x32_bf16(
            false, a2, false, b, (short)0, acc[2], false, false);
        acc[3] = __builtin_amdgcn_wmma_f32_16x16x32_bf16(
            false, a3, false, b, (short)0, acc[3], false, false);
    }
    const int col = n0 + row;
    const float bv = bias[col];
    #pragma unroll
    for (int t = 0; t < 4; ++t) {
        float* out = H + (size_t)(m0 + t * 16 + hi * 8) * HID + col;
        #pragma unroll
        for (int j = 0; j < 8; ++j) {             // D: VGPR j -> M = j + hi*8
            float v = acc[t][j] + bv;
            out[(size_t)j * HID] = v > 0.0f ? v : 0.0f;
        }
    }
}

// OUTP[1024,48] = A2[1024,512] @ W^T + b2 (padded cols get 0 weight/bias).
// Same 64x16 wave tile; 3 waves cover the 48 padded output columns.
__global__ __launch_bounds__(96) void gemm2_wmma(
    const __bf16* __restrict__ A, const __bf16* __restrict__ BT,
    const float* __restrict__ bias, float* __restrict__ OUTP) {
    const int lane = threadIdx.x & 31;
    const int wave = threadIdx.x >> 5;          // 3 waves -> 3 n-tiles
    const int m0 = blockIdx.x * 64;
    const int n0 = wave * 16;
    const int row = lane & 15;
    const int hi  = lane >> 4;
    const __bf16* aptr = A  + (size_t)(m0 + row) * K2 + hi * 8;
    const __bf16* bptr = BT + (size_t)(n0 + row) * K2 + hi * 8;
    const f32x8 z = {0.f, 0.f, 0.f, 0.f, 0.f, 0.f, 0.f, 0.f};
    f32x8 acc[4] = {z, z, z, z};
    for (int kk = 0; kk < K2 / 32; ++kk) {
        const int ko = kk * 32;
        bf16x16 b = load_frag(bptr + ko);
        bf16x16 a0 = load_frag(aptr + ko);
        bf16x16 a1 = load_frag(aptr + ko + (size_t)16 * K2);
        bf16x16 a2 = load_frag(aptr + ko + (size_t)32 * K2);
        bf16x16 a3 = load_frag(aptr + ko + (size_t)48 * K2);
        acc[0] = __builtin_amdgcn_wmma_f32_16x16x32_bf16(
            false, a0, false, b, (short)0, acc[0], false, false);
        acc[1] = __builtin_amdgcn_wmma_f32_16x16x32_bf16(
            false, a1, false, b, (short)0, acc[1], false, false);
        acc[2] = __builtin_amdgcn_wmma_f32_16x16x32_bf16(
            false, a2, false, b, (short)0, acc[2], false, false);
        acc[3] = __builtin_amdgcn_wmma_f32_16x16x32_bf16(
            false, a3, false, b, (short)0, acc[3], false, false);
    }
    const int col = n0 + row;
    const float bv = (col < OUTC) ? bias[col] : 0.0f;
    #pragma unroll
    for (int t = 0; t < 4; ++t) {
        float* out = OUTP + (size_t)(m0 + t * 16 + hi * 8) * NPAD2 + col;
        #pragma unroll
        for (int j = 0; j < 8; ++j)
            out[(size_t)j * NPAD2] = acc[t][j] + bv;
    }
}

// One wave per row: log_softmax over 41 valid columns of the padded buffer.
__global__ __launch_bounds__(32) void log_softmax_rows(
    const float* __restrict__ OUTP, float* __restrict__ out) {
    const int r = blockIdx.x;
    const int l = threadIdx.x;
    const float* row = OUTP + (size_t)r * NPAD2;
    float v0 = (l < OUTC)      ? row[l]      : -3.4e38f;
    float v1 = (l + 32 < OUTC) ? row[l + 32] : -3.4e38f;
    float m = fmaxf(v0, v1);
    #pragma unroll
    for (int o = 16; o > 0; o >>= 1) m = fmaxf(m, __shfl_xor(m, o, 32));
    float s = ((l < OUTC) ? __expf(v0 - m) : 0.0f) +
              ((l + 32 < OUTC) ? __expf(v1 - m) : 0.0f);
    #pragma unroll
    for (int o = 16; o > 0; o >>= 1) s += __shfl_xor(s, o, 32);
    float lse = m + __logf(s);
    if (l < OUTC)      out[(size_t)r * OUTC + l]      = v0 - lse;
    if (l + 32 < OUTC) out[(size_t)r * OUTC + l + 32] = v1 - lse;
}

// ---------------------------------------------------------------------------
// Workspace layout (bytes; all bf16 bases 16B-aligned for b128 fragment loads)
static constexpr size_t F_SUM1 = 0;
static constexpr size_t F_CNT1 = F_SUM1 + (size_t)GN1 * CIN;
static constexpr size_t F_SUM2 = F_CNT1 + GN1;
static constexpr size_t F_CNT2 = F_SUM2 + (size_t)GN0 * HID;
static constexpr size_t ZERO_FLOATS = F_CNT2 + GN0;            // 7,055,360
static constexpr size_t B_W1T = ZERO_FLOATS * 4;               // bytes
static constexpr size_t B_W2T = B_W1T + (size_t)HID * K1 * 2;
static constexpr size_t B_A1  = B_W2T + (size_t)NPAD2 * K2 * 2;
static constexpr size_t B_A2  = B_A1 + (size_t)GN1 * K1 * 2;
static constexpr size_t B_H   = B_A2 + (size_t)GN0 * K2 * 2;
static constexpr size_t B_OP  = B_H + (size_t)GN1 * HID * 4;   // end ~66 MB

extern "C" void kernel_launch(void* const* d_in, const int* in_sizes, int n_in,
                              void* d_out, int out_size, void* d_ws, size_t ws_size,
                              hipStream_t stream) {
    (void)in_sizes; (void)n_in; (void)out_size; (void)ws_size;
    const float* x    = (const float*)d_in[0];
    const int*   src1 = (const int*)d_in[1];
    const int*   dst1 = (const int*)d_in[2];
    const int*   src2 = (const int*)d_in[3];
    const int*   dst2 = (const int*)d_in[4];
    const float* W1l  = (const float*)d_in[5];
    const float* W1r  = (const float*)d_in[6];
    const float* b1   = (const float*)d_in[7];
    const float* W2l  = (const float*)d_in[8];
    const float* W2r  = (const float*)d_in[9];
    const float* b2   = (const float*)d_in[10];
    float* out = (float*)d_out;

    char* ws = (char*)d_ws;
    float*  sum1 = (float*)ws + F_SUM1;
    float*  cnt1 = (float*)ws + F_CNT1;
    float*  sum2 = (float*)ws + F_SUM2;
    float*  cnt2 = (float*)ws + F_CNT2;
    __bf16* W1T  = (__bf16*)(ws + B_W1T);
    __bf16* W2T  = (__bf16*)(ws + B_W2T);
    __bf16* A1   = (__bf16*)(ws + B_A1);
    __bf16* A2   = (__bf16*)(ws + B_A2);
    float*  H    = (float*)(ws + B_H);
    float*  OUTP = (float*)(ws + B_OP);

    // 1. zero the atomic accumulators (sum1|cnt1|sum2|cnt2 are contiguous)
    zero_f32<<<2048, 256, 0, stream>>>((float*)ws, (long)ZERO_FLOATS);
    // 2. pack transposed bf16 weights (stay L2-resident through the GEMMs)
    pack_w1<<<(HID * K1) / 256, 256, 0, stream>>>(W1l, W1r, W1T);
    pack_w2<<<(NPAD2 * K2) / 256, 256, 0, stream>>>(W2l, W2r, W2T);
    // 3. layer-1 segment sum + counts (memory/atomic bound: ~0.8 GB traffic)
    scatter_mean<<<GE1, 128, 0, stream>>>(x, src1, dst1, sum1, cnt1, CIN);
    // 4. A1 = [mean | x_dst | pad] in bf16
    build_a1<<<GN1, 256, 0, stream>>>(sum1, cnt1, x, A1);
    // 5. H = ReLU(A1 @ [W1l;W1r] + b1) on WMMA bf16, f32 accumulate
    gemm1_wmma<<<dim3(GN1 / 64, 4), 128, 0, stream>>>(A1, W1T, b1, H);
    // 6. layer-2 segment sum + counts
    scatter_mean<<<GE2, 128, 0, stream>>>(H, src2, dst2, sum2, cnt2, HID);
    // 7. A2 = [mean | h_dst] in bf16
    build_a2<<<GN0, 256, 0, stream>>>(sum2, cnt2, H, A2);
    // 8. OUTP = A2 @ [W2l;W2r] + b2 (48-col padded) on WMMA
    gemm2_wmma<<<GN0 / 64, 96, 0, stream>>>(A2, W2T, b2, OUTP);
    // 9. row-wise log_softmax over the 41 valid columns
    log_softmax_rows<<<GN0, 32, 0, stream>>>(OUTP, out);
}
